// EncoderText_44324062495190
// MI455X (gfx1250) — matrix operand order repaired and checked by reading, not compiled, not run on hardware
//
#include <hip/hip_runtime.h>

typedef __attribute__((ext_vector_type(16))) _Float16 v16h;
typedef __attribute__((ext_vector_type(8)))  float    v8f;

#define Bn 128
#define Ln 48
#define An 16
#define Dn 256
#define Hn 128
#define Sn 47
#define NEGC 1e10f

#define XS_STRIDE   260   // f32 rows, stride 260 dwords (mod 64 = 4): conflict-free 16-row reads
#define SYN_STRIDE  260
#define NSYN_STRIDE 264   // f16 rows, 132 dwords (mod 64 = 4)
#define TPB_TILES   4     // (b,l) tiles per persistent block in kernel 1

// Per-lane async copy: 16 bytes global -> LDS, tracked by ASYNCcnt (ISA 15.18.3 op 98).
__device__ __forceinline__ void async_b128_to_lds(uint32_t lds_off, const float* gaddr) {
  asm volatile("global_load_async_to_lds_b128 %0, %1, off"
               :: "v"(lds_off), "v"(gaddr) : "memory");
}

// ---------------------------------------------------------------------------
// Kernel 1: sem0[b,l,:] = max_a( x[b,l,a,:] @ w_sem^T - NEG*(1-mask) ),
//           out_word = sem0 * word_mask.
// Persistent: each block runs TPB_TILES (b,l) tiles with a double-buffered
// GLOBAL_LOAD_ASYNC_TO_LDS_B128 pipeline (memory-bound stage: overlap the
// 16KB tile fetch of t+1 with WMMA on t).  f16 WMMA, f32 accumulate.
// ---------------------------------------------------------------------------
__global__ __launch_bounds__(256) void sem_gemm_kernel(
    const float* __restrict__ x, const int* __restrict__ lengths,
    const float* __restrict__ amask, const float* __restrict__ w_sem,
    float* __restrict__ sem0, float* __restrict__ out_word)
{
  __shared__ __attribute__((aligned(16))) float xs[2][An][XS_STRIDE];
  __shared__ float am[2][An];
  const int tid = threadIdx.x;
  const int wave = tid >> 5, lane = tid & 31;
  const int hi = lane >> 4, ln = lane & 15;
  const int nt0 = wave * 2, nt1 = wave * 2 + 1;   // 16 col-tiles / 8 waves
  const int tile0 = blockIdx.x * TPB_TILES;

  // prologue: issue async copy of tile0 into buffer 0 (4 x b128 per thread)
  {
    const float* xp = x + (size_t)tile0 * (An * Dn);
    uint32_t lbase = (uint32_t)(uintptr_t)&xs[0][0][0];
#pragma unroll
    for (int i = 0; i < 4; ++i) {
      int e = tid * 4 + i * 1024;                 // 4 consecutive floats
      int a = e >> 8, d = e & 255;
      async_b128_to_lds(lbase + (uint32_t)(a * XS_STRIDE + d) * 4, xp + e);
    }
    if (tid < An) am[0][tid] = amask[(size_t)tile0 * An + tid];
  }

  for (int t = 0; t < TPB_TILES; ++t) {
    const int buf = t & 1;
    const int bl  = tile0 + t;                    // b*48 + l
    if (t + 1 < TPB_TILES) {
      // issue tile t+1 into the other buffer, then wait until only its
      // 4 ops remain outstanding => tile t's copies are complete (in-order).
      const float* xp = x + (size_t)(bl + 1) * (An * Dn);
      uint32_t lbase = (uint32_t)(uintptr_t)&xs[buf ^ 1][0][0];
#pragma unroll
      for (int i = 0; i < 4; ++i) {
        int e = tid * 4 + i * 1024;
        int a = e >> 8, d = e & 255;
        async_b128_to_lds(lbase + (uint32_t)(a * XS_STRIDE + d) * 4, xp + e);
      }
      if (tid < An) am[buf ^ 1][tid] = amask[(size_t)(bl + 1) * An + tid];
      asm volatile("s_wait_asynccnt 0x4" ::: "memory");
    } else {
      asm volatile("s_wait_asynccnt 0x0" ::: "memory");
    }
    __syncthreads();

    const int b = bl / Ln, l = bl % Ln;
    const float* xrow = &xs[buf][ln][0];
    v8f acc0 = {}; v8f acc1 = {};
    for (int k0 = 0; k0 < Dn; k0 += 32) {
      // A fragment: row = lane&15, contiguous k = 16*hi + e, f32->f16 on the fly
      v16h afrag;
#pragma unroll
      for (int e = 0; e < 16; ++e) {
        int v = e >> 1, p = e & 1;
        int kk = 16*hi + e; (void)v; (void)p;
        afrag[e] = (_Float16)xrow[k0 + kk];
      }
      // B fragment: B[k][n] = w_sem[n][k]; lane = col n, 16 contiguous K
      const float* wr0 = w_sem + (size_t)(nt0*16 + ln) * Dn + k0 + 16*hi;
      const float* wr1 = w_sem + (size_t)(nt1*16 + ln) * Dn + k0 + 16*hi;
      __builtin_prefetch(wr0 + 32, 0, 0);         // global_prefetch_b8 (L2-resident)
      v16h bf0, bf1;
#pragma unroll
      for (int e = 0; e < 16; ++e) { bf0[e] = (_Float16)wr0[e]; bf1[e] = (_Float16)wr1[e]; }
      acc0 = __builtin_amdgcn_wmma_f32_16x16x32_f16(false, afrag, false, bf0, (short)0, acc0, false, false);
      acc1 = __builtin_amdgcn_wmma_f32_16x16x32_f16(false, afrag, false, bf1, (short)0, acc1, false, false);
    }

    // C layout: VGPR e -> row e (lanes 0-15) / row e+8 (lanes 16-31), col = lane&15
    float m0 = -3.0e38f, m1 = -3.0e38f;
#pragma unroll
    for (int e = 0; e < 8; ++e) {
      float pen = NEGC * (1.0f - am[buf][e + 8*hi]);
      m0 = fmaxf(m0, acc0[e] - pen);
      m1 = fmaxf(m1, acc1[e] - pen);
    }
    m0 = fmaxf(m0, __shfl_xor(m0, 16, 32));
    m1 = fmaxf(m1, __shfl_xor(m1, 16, 32));
    if (hi == 0) {
      size_t base = (size_t)bl * Dn;
      float wm = (l < lengths[b]) ? 1.0f : 0.0f;
      sem0[base + nt0*16 + ln]     = m0;
      sem0[base + nt1*16 + ln]     = m1;
      out_word[base + nt0*16 + ln] = m0 * wm;
      out_word[base + nt1*16 + ln] = m1 * wm;
    }
    __syncthreads();   // everyone done reading xs[buf] before it is re-issued
  }
}

// ---------------------------------------------------------------------------
// Kernel 2: full 47-step greedy merge scan. One block (8 waves) per batch b.
// syn/sem live in LDS; merges use a perm[] indirection (O(1) data movement).
// Per-step GEMM M=48,N=128,K=512 -> wave w owns N-tile w, 3 M-tiles, f16 WMMA.
// ---------------------------------------------------------------------------
__global__ __launch_bounds__(256) void scan_kernel(
    const float* __restrict__ sem0, const int* __restrict__ lengths,
    const float* __restrict__ w1, const float* __restrict__ b1,
    const float* __restrict__ w2,
    float* __restrict__ feat_out, float* __restrict__ left_out,
    float* __restrict__ right_out, float* __restrict__ idx_out,
    float* __restrict__ tp_out, float* __restrict__ span_out)
{
  extern __shared__ unsigned char smem_raw[];
  float*    syn   = (float*)smem_raw;                       // 48 x SYN_STRIDE
  float*    semA  = syn + Ln * SYN_STRIDE;                  // 48 x SYN_STRIDE
  _Float16* nsyn  = (_Float16*)(semA + Ln * SYN_STRIDE);    // 48 x NSYN_STRIDE (l2norm'd syn, f16)
  float*    fpool = (float*)(nsyn + Ln * NSYN_STRIDE);
  float* invn   = fpool; fpool += Ln;
  float* logits = fpool; fpool += Ln;
  float* b1s    = fpool; fpool += Hn;
  float* w2s    = fpool; fpool += Hn;
  float* csyn   = fpool; fpool += Dn;
  float* csem   = fpool; fpool += Dn;
  float* redv   = fpool; fpool += 4;
  float* s_tp   = fpool; fpool += 1;
  int* ipool = (int*)fpool;
  int* lbp  = ipool; ipool += Ln;   // left bound per *physical* slot
  int* rbp  = ipool; ipool += Ln;   // right bound per *physical* slot
  int* perm = ipool; ipool += Ln;   // logical row -> physical slot
  int* s_idx = ipool;

  const int b = blockIdx.x, tid = threadIdx.x;
  const int wave = tid >> 5, lane = tid & 31;
  const int hi = lane >> 4, ln = lane & 15;
  const int len = lengths[b];

  const float* s0 = sem0 + (size_t)b * Ln * Dn;
  for (int e = tid; e < Ln * Dn; e += 256) {
    int r = e >> 8, d = e & 255;
    float v = s0[e];
    syn[r * SYN_STRIDE + d] = v;
    semA[r * SYN_STRIDE + d] = v;
  }
  if (tid < Hn) { b1s[tid] = b1[tid]; w2s[tid] = w2[tid]; }
  if (tid < Ln) { lbp[tid] = tid; rbp[tid] = tid; perm[tid] = tid; }
  __syncthreads();

  for (int step = 0; step < Sn; ++step) {
    // ---- 1/l2norm of every physical syn row (wave-parallel) ----
    for (int r = wave; r < Ln; r += 8) {
      float s = 0.0f;
      for (int d = lane; d < Dn; d += 32) { float v = syn[r * SYN_STRIDE + d]; s += v * v; }
#pragma unroll
      for (int off = 16; off > 0; off >>= 1) s += __shfl_xor(s, off, 32);
      if (lane == 0) invn[r] = 1.0f / (sqrtf(s) + 1e-8f);
    }
    if (tid < Ln) logits[tid] = 0.0f;
    if (tid < 4)  redv[tid] = 0.0f;
    __syncthreads();
    for (int e = tid; e < Ln * Dn; e += 256) {
      int r = e >> 8, d = e & 255;
      nsyn[r * NSYN_STRIDE + d] = (_Float16)(syn[r * SYN_STRIDE + d] * invn[r]);
    }
    __syncthreads();

    // ---- GEMM: h = relu(feats @ w1^T + b1); logits[j] += h[j,:]·w2 ----
    // feats[j] = [ nsyn[perm[j+1]] , nsyn[perm[j]] ]  (K = 512)
    const int nt = wave;
    int pj[3], pj1[3];
#pragma unroll
    for (int mt = 0; mt < 3; ++mt) {
      int j = mt * 16 + ln;
      pj[mt]  = (j < Sn) ? perm[j]     : 0;   // j=47 pad row: finite garbage, masked later
      pj1[mt] = (j < Sn) ? perm[j + 1] : 0;
    }
    v8f acc[3] = { {}, {}, {} };
    for (int k0 = 0; k0 < 512; k0 += 32) {
      const float* wr = w1 + (size_t)(nt*16 + ln) * 512 + k0 + 16*hi;
      __builtin_prefetch(wr + 32, 0, 0);
      v16h bfrag;
#pragma unroll
      for (int e = 0; e < 16; ++e) bfrag[e] = (_Float16)wr[e];
#pragma unroll
      for (int mt = 0; mt < 3; ++mt) {
        const _Float16* rowp = (k0 < 256)
            ? (nsyn + pj1[mt] * NSYN_STRIDE + k0)
            : (nsyn + pj[mt]  * NSYN_STRIDE + (k0 - 256));
        v16h afrag;
#pragma unroll
        for (int e = 0; e < 16; ++e) {
          int v = e >> 1, p = e & 1;
          int kk = 16*hi + e; (void)v; (void)p;
          afrag[e] = rowp[kk];
        }
        acc[mt] = __builtin_amdgcn_wmma_f32_16x16x32_f16(false, afrag, false, bfrag,
                                                         (short)0, acc[mt], false, false);
      }
    }
    // relu + dot with w2, reduce the 16 lanes of each half (same row j)
#pragma unroll
    for (int mt = 0; mt < 3; ++mt) {
#pragma unroll
      for (int e = 0; e < 8; ++e) {
        int j = mt * 16 + e + 8 * hi;
        int col = nt * 16 + ln;
        float h = fmaxf(acc[mt][e] + b1s[col], 0.0f);
        float v = h * w2s[col];
        v += __shfl_xor(v, 1, 32); v += __shfl_xor(v, 2, 32);
        v += __shfl_xor(v, 4, 32); v += __shfl_xor(v, 8, 32);
        if (ln == 0 && j < Sn) atomicAdd(&logits[j], v);
      }
    }
    __syncthreads();

    // ---- masked argmax + softmax prob at argmax (wave 0) ----
    int valid = len - 1 - step; if (valid < 0) valid = 0;
    if (wave == 0) {
      float v0 = -3.4e38f; int i0 = 0;
      for (int j = lane; j < Sn; j += 32) {
        float lv = logits[j] - ((j >= valid) ? NEGC : 0.0f);
        if (lv > v0) { v0 = lv; i0 = j; }            // ascending j -> first max kept
      }
#pragma unroll
      for (int off = 16; off > 0; off >>= 1) {
        float ov = __shfl_xor(v0, off, 32); int oi = __shfl_xor(i0, off, 32);
        if (ov > v0 || (ov == v0 && oi < i0)) { v0 = ov; i0 = oi; }
      }
      float s = 0.0f;
      for (int j = lane; j < Sn; j += 32) {
        float lv = logits[j] - ((j >= valid) ? NEGC : 0.0f);
        s += __expf(lv - v0);
      }
#pragma unroll
      for (int off = 16; off > 0; off >>= 1) s += __shfl_xor(s, off, 32);
      if (lane == 0) { *s_idx = i0; *s_tp = 1.0f / s; }   // probs[idx] = 1/sum
    }
    __syncthreads();

    // ---- gather, norms, outputs ----
    const int idx = *s_idx;
    const int pl = perm[idx], pr = perm[idx + 1];
    const int d = tid;                    // 256 threads == D
    float sl = semA[pl * SYN_STRIDE + d], sr = semA[pr * SYN_STRIDE + d];
    float yl = syn[pl * SYN_STRIDE + d],  yr = syn[pr * SYN_STRIDE + d];
    {
      float q0 = sl*sl, q1 = sr*sr, q2 = (sl+sr)*(sl+sr), q3 = (yl+yr)*(yl+yr);
#pragma unroll
      for (int off = 16; off > 0; off >>= 1) {
        q0 += __shfl_xor(q0, off, 32); q1 += __shfl_xor(q1, off, 32);
        q2 += __shfl_xor(q2, off, 32); q3 += __shfl_xor(q3, off, 32);
      }
      if (lane == 0) {
        atomicAdd(&redv[0], q0); atomicAdd(&redv[1], q1);
        atomicAdd(&redv[2], q2); atomicAdd(&redv[3], q3);
      }
    }
    __syncthreads();
    {
      float il = 1.0f / (sqrtf(redv[0]) + 1e-8f);
      float ir = 1.0f / (sqrtf(redv[1]) + 1e-8f);
      float ic = 1.0f / (sqrtf(redv[2]) + 1e-8f);
      float iy = 1.0f / (sqrtf(redv[3]) + 1e-8f);
      float undone = (valid > 0) ? 1.0f : 0.0f;
      float fc = (sl + sr) * ic;
      size_t o = ((size_t)step * Bn + b) * Dn + d;
      feat_out[o]  = fc * undone;
      left_out[o]  = sl * il * undone;
      right_out[o] = sr * ir * undone;
      csem[d] = fc;
      csyn[d] = (yl + yr) * iy;
      if (tid == 0) {
        size_t so = (size_t)step * Bn + b;
        idx_out[so] = (float)idx;
        tp_out[so]  = *s_tp;
        span_out[so * 2 + 0] = (float)lbp[pl];
        span_out[so * 2 + 1] = (float)rbp[pr];
      }
    }
    __syncthreads();

    // ---- merge: write center into slot pl, drop logical row idx+1 via perm shift ----
    syn[pl * SYN_STRIDE + d]  = csyn[d];
    semA[pl * SYN_STRIDE + d] = csem[d];
    if (tid == 0) {
      rbp[pl] = rbp[pr];                       // merged span: [lbp[pl], rbp[pr]]
      for (int j = idx + 1; j < Sn; ++j) perm[j] = perm[j + 1];
    }
    __syncthreads();
  }
}

// ---------------------------------------------------------------------------
extern "C" void kernel_launch(void* const* d_in, const int* in_sizes, int n_in,
                              void* d_out, int out_size, void* d_ws, size_t ws_size,
                              hipStream_t stream) {
  (void)in_sizes; (void)n_in; (void)out_size; (void)ws_size;
  const float* x       = (const float*)d_in[0];
  const int*   lengths = (const int*)d_in[1];
  const float* amask   = (const float*)d_in[2];
  const float* w_sem   = (const float*)d_in[3];
  const float* w1      = (const float*)d_in[4];
  const float* b1      = (const float*)d_in[5];
  const float* w2      = (const float*)d_in[6];

  float* out = (float*)d_out;
  // output tuple, flattened in return order
  float* feat_out  = out;                                    // [47,128,256]
  float* left_out  = out + (size_t)Sn * Bn * Dn;             // [47,128,256]
  float* right_out = out + (size_t)2 * Sn * Bn * Dn;         // [47,128,256]
  float* out_word  = out + (size_t)3 * Sn * Bn * Dn;         // [128,48,256]
  float* idx_out   = out_word + (size_t)Bn * Ln * Dn;        // [47,128]
  float* tp_out    = idx_out + (size_t)Sn * Bn;              // [47,128]
  float* span_out  = tp_out + (size_t)Sn * Bn;               // [47,128,2]

  float* sem0 = (float*)d_ws;                                // [128,48,256]

  sem_gemm_kernel<<<(Bn * Ln) / TPB_TILES, 256, 0, stream>>>(
      x, lengths, amask, w_sem, sem0, out_word);

  size_t smem2 = (size_t)2 * Ln * SYN_STRIDE * 4            // syn + sem (f32)
               + (size_t)Ln * NSYN_STRIDE * 2               // nsyn (f16)
               + (size_t)(Ln + Ln + Hn + Hn + Dn + Dn + 4 + 1) * 4   // float pool
               + (size_t)(3 * Ln + 1) * 4;                  // int pool
  scan_kernel<<<Bn, 256, smem2, stream>>>(sem0, lengths, w1, b1, w2,
                                          feat_out, left_out, right_out,
                                          idx_out, tp_out, span_out);
}
